// CrossAttention_49125835931836
// MI455X (gfx1250) — compile-verified, matrix operand order checked by reading, not run on hardware
//
#include <hip/hip_runtime.h>
#include <hip/hip_bf16.h>
#include <math.h>

#define DEV __device__ __forceinline__

typedef __attribute__((ext_vector_type(16))) __bf16 v16bf;
typedef __attribute__((ext_vector_type(8)))  float  v8f;
typedef __attribute__((ext_vector_type(8)))  unsigned short us8;

DEV unsigned short f2bf(float f) {
  unsigned int u = __float_as_uint(f);
  u += 0x7FFFu + ((u >> 16) & 1u);   // round-to-nearest-even
  return (unsigned short)(u >> 16);
}

// Low 32 bits of a flat pointer into LDS == LDS byte offset (aperture rule).
DEV unsigned lds_off_of(const void* p) { return (unsigned)(unsigned long long)p; }

// CDNA5 async copy: 16B global -> LDS per lane, tracked by ASYNCcnt.
DEV void async_b128(unsigned lds_byte, unsigned gv_byte, const void* sbase) {
  asm volatile("global_load_async_to_lds_b128 %0, %1, %2"
               :: "v"(lds_byte), "v"(gv_byte),
                  "s"((unsigned long long)sbase)
               : "memory");
}
DEV void wait_async0() { asm volatile("s_wait_asynccnt 0x0" ::: "memory"); }
DEV void wait_async4() { asm volatile("s_wait_asynccnt 0x4" ::: "memory"); }

// Load one 16x32 bf16 fragment (A-layout; B identical with lane = out column,
// since B = W^T and W rows are stored [n][K]). rp -> tile[lane&15] + kbase.
DEV v16bf ld_frag(const unsigned short* rp, int h) {
  union { v16bf v; us8 u[2]; } f;
  f.u[0] = *(const us8*)(rp + 8 * h);        // K = 8h .. 8h+7
  f.u[1] = *(const us8*)(rp + 16 + 8 * h);   // K = 16+8h .. 16+8h+7
  return f.v;
}

DEV v8f wmma_bf16(v16bf a, v16bf b, v8f c) {
  return __builtin_amdgcn_wmma_f32_16x16x32_bf16(false, a, false, b, (short)0, c,
                                                 false, false);
}

DEV v8f v8f_zero() { v8f z = {0.f,0.f,0.f,0.f,0.f,0.f,0.f,0.f}; return z; }

// ---------------------------------------------------------------------------
// f32 -> bf16 convert (weights, once per launch).
// ---------------------------------------------------------------------------
__global__ __launch_bounds__(256)
void cvt_kernel(const float* __restrict__ S, unsigned short* __restrict__ D, int n) {
  int i = blockIdx.x * 1024 + threadIdx.x;
  for (int k = 0; k < 4; ++k, i += 256)
    if (i < n) D[i] = f2bf(S[i]);
}

// ---------------------------------------------------------------------------
// LayerNorm over last dim = 768; f32 in, bf16 out. One block per row.
// ---------------------------------------------------------------------------
__global__ __launch_bounds__(256)
void ln_kernel(const float* __restrict__ X, const float* __restrict__ g,
               const float* __restrict__ b, unsigned short* __restrict__ Y) {
  const int row = blockIdx.x;
  const float* xr = X + (size_t)row * 768;
  float v[3], s = 0.f, s2 = 0.f;
  for (int i = 0; i < 3; ++i) {
    v[i] = xr[threadIdx.x + 256 * i];
    s += v[i]; s2 += v[i] * v[i];
  }
  for (int m = 16; m >= 1; m >>= 1) {
    s  += __shfl_xor(s,  m, 32);
    s2 += __shfl_xor(s2, m, 32);
  }
  __shared__ float red[16];
  const int wave = threadIdx.x >> 5, lane = threadIdx.x & 31;
  if (lane == 0) { red[wave] = s; red[8 + wave] = s2; }
  __syncthreads();
  s = 0.f; s2 = 0.f;
  for (int w = 0; w < 8; ++w) { s += red[w]; s2 += red[8 + w]; }
  const float mean = s * (1.0f / 768.0f);
  const float var  = s2 * (1.0f / 768.0f) - mean * mean;
  const float inv  = rsqrtf(var + 1e-5f);
  unsigned short* yr = Y + (size_t)row * 768;
  for (int i = 0; i < 3; ++i) {
    const int c = threadIdx.x + 256 * i;
    yr[c] = f2bf((v[i] - mean) * inv * g[c] + b[c]);
  }
}

// ---------------------------------------------------------------------------
// GEMM: C[M,N] = A[M,K](bf16) * W[N,K](bf16)^T + bias(f32), epilogue:
//   EPI 0: +bias   EPI 1: +bias +R(f32)   EPI 2: gelu(+bias)
//   OUTB 1: bf16 out   OUTB 0: f32 out
// 128x128 block tile, BK=32, 8 waves, wave tile 32x64 (2x4 WMMA accums).
// Double-buffered async global->LDS staging: tile t+1 copies while tile t
// runs on the WMMA pipe (per-wave s_wait_asynccnt 4 = oldest tile complete).
// ---------------------------------------------------------------------------
template <int EPI, int OUTB>
__global__ __launch_bounds__(256)
void gemm_kernel(const unsigned short* __restrict__ A,
                 const unsigned short* __restrict__ W,
                 const float* __restrict__ bias, const float* __restrict__ R,
                 void* __restrict__ Cout, int M, int N, int K) {
  __shared__ __align__(16) unsigned short sA[2][128 * 32];
  __shared__ __align__(16) unsigned short sB[2][128 * 32];
  const unsigned ldsA[2] = { lds_off_of(sA[0]), lds_off_of(sA[1]) };
  const unsigned ldsB[2] = { lds_off_of(sB[0]), lds_off_of(sB[1]) };
  const int tid = threadIdx.x;
  const int lane = tid & 31, wave = tid >> 5;
  const int h = lane >> 4, n15 = lane & 15;
  const int bm = blockIdx.y * 128;
  const int bn = blockIdx.x * 128;
  const int wm = (wave & 3) * 32;
  const int wn = (wave >> 2) * 64;

  // This thread's two 16B chunks per 128x32 tile: rows r0 and r0+64.
  const int r0 = tid >> 2;
  const unsigned q0  = (tid & 3) * 16;
  const unsigned gv0 = (unsigned)(r0 * K) * 2u + q0;
  const unsigned gv1 = gv0 + (unsigned)(64 * K) * 2u;
  const unsigned lo0 = (unsigned)(r0 * 64) + q0;
  const unsigned lo1 = lo0 + 64 * 64;

  const char* aRow = (const char*)(A + (size_t)bm * K);
  const char* bRow = (const char*)(W + (size_t)bn * K);
  auto stage = [&](int kk, int bsel) {
    const char* aK = aRow + (size_t)kk * 2;
    const char* bK = bRow + (size_t)kk * 2;
    async_b128(ldsA[bsel] + lo0, gv0, aK);
    async_b128(ldsA[bsel] + lo1, gv1, aK);
    async_b128(ldsB[bsel] + lo0, gv0, bK);
    async_b128(ldsB[bsel] + lo1, gv1, bK);
  };

  v8f acc[2][4];
  for (int i = 0; i < 2; ++i)
    for (int j = 0; j < 4; ++j) acc[i][j] = v8f_zero();

  const int NT = K >> 5;
  stage(0, 0);
  for (int t = 0; t < NT; ++t) {
    const int cur = t & 1;
    if (t + 1 < NT) { stage((t + 1) << 5, cur ^ 1); wait_async4(); }
    else            { wait_async0(); }
    __syncthreads();
    v16bf afr[2], bfr[4];
    for (int i = 0; i < 2; ++i) afr[i] = ld_frag(&sA[cur][(wm + i * 16 + n15) * 32], h);
    for (int j = 0; j < 4; ++j) bfr[j] = ld_frag(&sB[cur][(wn + j * 16 + n15) * 32], h);
    for (int i = 0; i < 2; ++i)
      for (int j = 0; j < 4; ++j) acc[i][j] = wmma_bf16(afr[i], bfr[j], acc[i][j]);
    __syncthreads();   // all reads of buf[cur] done before it is re-staged
  }

  float* Cf = (float*)Cout;
  unsigned short* Cb = (unsigned short*)Cout;
  for (int i = 0; i < 2; ++i)
    for (int j = 0; j < 4; ++j) {
      const int gn = bn + wn + j * 16 + n15;
      const float bs = bias[gn];
      unsigned idx = (unsigned)(bm + wm + i * 16 + 8 * h) * (unsigned)N + (unsigned)gn;
      for (int r = 0; r < 8; ++r, idx += (unsigned)N) {
        float vv = acc[i][j][r] + bs;
        if (EPI == 2) vv = 0.5f * vv * (1.0f + erff(vv * 0.70710678118654752f));
        if (EPI == 1) vv += R[idx];
        if (OUTB) Cb[idx] = f2bf(vv);
        else      Cf[idx] = vv;
      }
    }
}

// ---------------------------------------------------------------------------
// Fused biased-softmax cross attention, one (batch, head) per block.
//  Q: (B,LQ,768) bf16; KV: (B,LK,1536) bf16; O: (B,LQ,768) bf16.
//  S = QK^T*0.125 + rpb[relidx][hd]; P = softmax(S); O = P V.
// ---------------------------------------------------------------------------
template <int LQ, int LK, int WGM, int WGN, int ROFF, int QSH, int KSH,
          int OWM, int OWN>
__global__ __launch_bounds__(256)
void attn_kernel(const unsigned short* __restrict__ Q,
                 const unsigned short* __restrict__ KV,
                 const float* __restrict__ rpb, unsigned short* __restrict__ O) {
  constexpr int SM  = LQ / WGM / 16;
  constexpr int SN  = LK / WGN / 16;
  constexpr int OSM = LQ / OWM / 16;
  constexpr int OSN = 64 / OWN / 16;

  const int blk = blockIdx.x;
  const int b = blk / 12, hd = blk % 12;
  const unsigned short* qp = Q  + (size_t)b * LQ * 768  + (size_t)hd * 64;
  const unsigned short* kp = KV + (size_t)b * LK * 1536 + (size_t)hd * 64;
  const unsigned short* vp = kp + 768;

  struct QK { unsigned short q[LQ * 64]; unsigned short k[LK * 64]; };
  union SMem { QK qk; unsigned short p[LQ * LK]; };
  __shared__ __align__(16) SMem sm;                 // 40KB; P overlays dead Q/K
  __shared__ __align__(16) unsigned short sVt[64 * 32];
  __shared__ float sRed[WGN * LQ];

  const int tid = threadIdx.x;
  const int lane = tid & 31, wave = tid >> 5;
  const int h = lane >> 4, n15 = lane & 15;
  const unsigned ldsQ = lds_off_of(sm.qk.q);
  const unsigned ldsK = lds_off_of(sm.qk.k);

  // Async-stage Q and K (bf16, 16B chunks; 8 chunks per 64-elem row).
  for (int c = tid; c < LQ * 8; c += 256)
    async_b128(ldsQ + (unsigned)c * 16,
               (unsigned)((c >> 3) * 1536 + (c & 7) * 16), qp);
  for (int c = tid; c < LK * 8; c += 256)
    async_b128(ldsK + (unsigned)c * 16,
               (unsigned)((c >> 3) * 3072 + (c & 7) * 16), kp);
  wait_async0();
  __syncthreads();

  // ---- S = Q K^T (registers) ----
  const int wm = (wave % WGM) * (SM * 16);
  const int wn = (wave / WGM) * (SN * 16);
  v8f acc[SM][SN];
  for (int i = 0; i < SM; ++i)
    for (int j = 0; j < SN; ++j) acc[i][j] = v8f_zero();
  for (int kk = 0; kk < 64; kk += 32) {
    v16bf afr[SM], bfr[SN];
    for (int i = 0; i < SM; ++i) afr[i] = ld_frag(&sm.qk.q[(wm + i * 16 + n15) * 64 + kk], h);
    for (int j = 0; j < SN; ++j) bfr[j] = ld_frag(&sm.qk.k[(wn + j * 16 + n15) * 64 + kk], h);
    for (int i = 0; i < SM; ++i)
      for (int j = 0; j < SN; ++j) acc[i][j] = wmma_bf16(afr[i], bfr[j], acc[i][j]);
  }

  // ---- scale + relative position bias ----
  float sv[SM][SN][8];
  for (int i = 0; i < SM; ++i)
    for (int j = 0; j < SN; ++j)
      for (int r = 0; r < 8; ++r) {
        const int lq = wm + i * 16 + r + 8 * h;
        const int lk = wn + j * 16 + n15;
        const int r0 = (lq >> QSH) - (lk >> KSH) + ROFF;
        const int r1 = (lq & ((1 << QSH) - 1)) - (lk & ((1 << KSH) - 1)) + ROFF;
        sv[i][j][r] = acc[i][j][r] * 0.125f + rpb[(r0 * 23 + r1) * 12 + hd];
      }

  // ---- softmax over LK ----
  float rmax[SM][8], rsum[SM][8];
  for (int i = 0; i < SM; ++i)
    for (int r = 0; r < 8; ++r) {
      float m = -3.0e38f;
      for (int j = 0; j < SN; ++j) m = fmaxf(m, sv[i][j][r]);
      for (int msk = 1; msk <= 8; msk <<= 1) m = fmaxf(m, __shfl_xor(m, msk, 32));
      rmax[i][r] = m;
    }
  if (n15 == 0)
    for (int i = 0; i < SM; ++i)
      for (int r = 0; r < 8; ++r)
        sRed[(wave / WGM) * LQ + wm + i * 16 + r + 8 * h] = rmax[i][r];
  __syncthreads();
  for (int i = 0; i < SM; ++i)
    for (int r = 0; r < 8; ++r) {
      const int lq = wm + i * 16 + r + 8 * h;
      float m = sRed[lq];
      for (int w = 1; w < WGN; ++w) m = fmaxf(m, sRed[w * LQ + lq]);
      rmax[i][r] = m;
    }
  __syncthreads();
  for (int i = 0; i < SM; ++i)
    for (int r = 0; r < 8; ++r) {
      float s = 0.f;
      for (int j = 0; j < SN; ++j) {
        sv[i][j][r] = __expf(sv[i][j][r] - rmax[i][r]);
        s += sv[i][j][r];
      }
      for (int msk = 1; msk <= 8; msk <<= 1) s += __shfl_xor(s, msk, 32);
      rsum[i][r] = s;
    }
  if (n15 == 0)
    for (int i = 0; i < SM; ++i)
      for (int r = 0; r < 8; ++r)
        sRed[(wave / WGM) * LQ + wm + i * 16 + r + 8 * h] = rsum[i][r];
  __syncthreads();
  for (int i = 0; i < SM; ++i)
    for (int r = 0; r < 8; ++r) {
      const int lq = wm + i * 16 + r + 8 * h;
      float s = sRed[lq];
      for (int w = 1; w < WGN; ++w) s += sRed[w * LQ + lq];
      rsum[i][r] = 1.0f / s;
    }
  // P -> LDS bf16 (overlays Q/K; all Q/K reads completed before barriers above)
  for (int i = 0; i < SM; ++i)
    for (int j = 0; j < SN; ++j)
      for (int r = 0; r < 8; ++r) {
        const int lq = wm + i * 16 + r + 8 * h;
        const int lk = wn + j * 16 + n15;
        sm.p[lq * LK + lk] = f2bf(sv[i][j][r] * rsum[i][r]);
      }
  __syncthreads();

  // ---- O = P V; V^T tiles staged with one b128 global load per thread ----
  const int om = (wave % OWM) * (OSM * 16);
  const int on = (wave / OWM) * (OSN * 16);
  v8f oacc[OSM][OSN];
  for (int i = 0; i < OSM; ++i)
    for (int j = 0; j < OSN; ++j) oacc[i][j] = v8f_zero();
  const int vl = tid >> 3;            // lk within tile: 0..31
  const int vd = (tid & 7) * 8;       // 8 consecutive d values
  for (int kk = 0; kk < LK; kk += 32) {
    const us8 vrow = *(const us8*)(vp + (unsigned)((kk + vl) * 1536) + vd);
    for (int e = 0; e < 8; ++e) sVt[(vd + e) * 32 + vl] = vrow[e];
    __syncthreads();
    v16bf afr[OSM], bfr[OSN];
    for (int i = 0; i < OSM; ++i) afr[i] = ld_frag(&sm.p[(om + i * 16 + n15) * LK + kk], h);
    for (int j = 0; j < OSN; ++j) bfr[j] = ld_frag(&sVt[(on + j * 16 + n15) * 32], h);
    for (int i = 0; i < OSM; ++i)
      for (int j = 0; j < OSN; ++j) oacc[i][j] = wmma_bf16(afr[i], bfr[j], oacc[i][j]);
    __syncthreads();
  }
  unsigned short* op = O + (size_t)b * LQ * 768 + (size_t)hd * 64;
  for (int i = 0; i < OSM; ++i)
    for (int j = 0; j < OSN; ++j) {
      unsigned idx = (unsigned)(om + i * 16 + 8 * h) * 768u + (unsigned)(on + j * 16 + n15);
      for (int r = 0; r < 8; ++r, idx += 768u) op[idx] = f2bf(oacc[i][j][r]);
    }
}

// ---------------------------------------------------------------------------
extern "C" void kernel_launch(void* const* d_in, const int* in_sizes, int n_in,
                              void* d_out, int out_size, void* d_ws, size_t ws_size,
                              hipStream_t stream) {
  (void)in_sizes; (void)n_in; (void)out_size; (void)ws_size;
  const float* z       = (const float*)d_in[0];
  const float* x       = (const float*)d_in[1];
  const float* z_ln1_g = (const float*)d_in[2];
  const float* z_ln1_b = (const float*)d_in[3];
  const float* x_ln1_g = (const float*)d_in[4];
  const float* x_ln1_b = (const float*)d_in[5];
  const float* zx_qw   = (const float*)d_in[6];
  const float* zx_qb   = (const float*)d_in[7];
  const float* zx_kvw  = (const float*)d_in[8];
  const float* zx_kvb  = (const float*)d_in[9];
  const float* zx_pw   = (const float*)d_in[10];
  const float* zx_pb   = (const float*)d_in[11];
  const float* zx_rpb  = (const float*)d_in[12];
  const float* xz_qw   = (const float*)d_in[13];
  const float* xz_qb   = (const float*)d_in[14];
  const float* xz_kvw  = (const float*)d_in[15];
  const float* xz_kvb  = (const float*)d_in[16];
  const float* xz_pw   = (const float*)d_in[17];
  const float* xz_pb   = (const float*)d_in[18];
  const float* xz_rpb  = (const float*)d_in[19];
  const float* z_ln2_g = (const float*)d_in[20];
  const float* z_ln2_b = (const float*)d_in[21];
  const float* z_fc1_w = (const float*)d_in[22];
  const float* z_fc1_b = (const float*)d_in[23];
  const float* z_fc2_w = (const float*)d_in[24];
  const float* z_fc2_b = (const float*)d_in[25];
  const float* x_ln2_g = (const float*)d_in[26];
  const float* x_ln2_b = (const float*)d_in[27];
  const float* x_fc1_w = (const float*)d_in[28];
  const float* x_fc1_b = (const float*)d_in[29];
  const float* x_fc2_w = (const float*)d_in[30];
  const float* x_fc2_b = (const float*)d_in[31];

  const size_t ZT = 64 * 64, XT = 64 * 256;
  typedef unsigned short u16;
  u16* ws = (u16*)d_ws;
  size_t off = 0;
  auto alloc = [&](size_t n) { u16* p = ws + off; off += n; return p; };
  // bf16 weight copies (persistent for the whole launch)
  u16* wb_zx_qw  = alloc(768 * 768);
  u16* wb_zx_kvw = alloc(1536 * 768);
  u16* wb_zx_pw  = alloc(768 * 768);
  u16* wb_xz_qw  = alloc(768 * 768);
  u16* wb_xz_kvw = alloc(1536 * 768);
  u16* wb_xz_pw  = alloc(768 * 768);
  u16* wb_z_fc1  = alloc(3072 * 768);
  u16* wb_z_fc2  = alloc(768 * 3072);
  u16* wb_x_fc1  = alloc(3072 * 768);
  u16* wb_x_fc2  = alloc(768 * 3072);
  // Phase-1 activations (bf16)
  u16* act = ws + off;
  u16* zn    = alloc(ZT * 768);
  u16* xn    = alloc(XT * 768);
  u16* zx_q  = alloc(ZT * 768);
  u16* zx_kv = alloc(XT * 1536);
  u16* xz_q  = alloc(XT * 768);
  u16* xz_kv = alloc(ZT * 1536);
  u16* zx_o  = alloc(ZT * 768);
  u16* xz_o  = alloc(XT * 768);
  // Phase-2 overlays (phase-1 activations dead by then)
  u16* z2n = act;
  u16* x2n = z2n + ZT * 768;
  u16* zh  = x2n + XT * 768;
  u16* xh  = zh + ZT * 3072;
  // d_out doubles as the f32 z2/x2 residual carrier
  float* z2 = (float*)d_out;
  float* x2 = z2 + ZT * 768;

  // 0) weights -> bf16
  auto cvt = [&](const float* s, u16* d, int n) {
    cvt_kernel<<<(n + 1023) / 1024, 256, 0, stream>>>(s, d, n);
  };
  cvt(zx_qw, wb_zx_qw, 768 * 768);    cvt(zx_kvw, wb_zx_kvw, 1536 * 768);
  cvt(zx_pw, wb_zx_pw, 768 * 768);    cvt(xz_qw, wb_xz_qw, 768 * 768);
  cvt(xz_kvw, wb_xz_kvw, 1536 * 768); cvt(xz_pw, wb_xz_pw, 768 * 768);
  cvt(z_fc1_w, wb_z_fc1, 3072 * 768); cvt(z_fc2_w, wb_z_fc2, 768 * 3072);
  cvt(x_fc1_w, wb_x_fc1, 3072 * 768); cvt(x_fc2_w, wb_x_fc2, 768 * 3072);

  // 1) LayerNorms (f32 -> bf16)
  ln_kernel<<<(int)ZT, 256, 0, stream>>>(z, z_ln1_g, z_ln1_b, zn);
  ln_kernel<<<(int)XT, 256, 0, stream>>>(x, x_ln1_g, x_ln1_b, xn);
  // 2) Projections (bf16 out)
  gemm_kernel<0,1><<<dim3(6,  ZT/128), 256, 0, stream>>>(zn, wb_zx_qw,  zx_qb,  nullptr, zx_q,  ZT, 768,  768);
  gemm_kernel<0,1><<<dim3(12, XT/128), 256, 0, stream>>>(xn, wb_zx_kvw, zx_kvb, nullptr, zx_kv, XT, 1536, 768);
  gemm_kernel<0,1><<<dim3(6,  XT/128), 256, 0, stream>>>(xn, wb_xz_qw,  xz_qb,  nullptr, xz_q,  XT, 768,  768);
  gemm_kernel<0,1><<<dim3(12, ZT/128), 256, 0, stream>>>(zn, wb_xz_kvw, xz_kvb, nullptr, xz_kv, ZT, 1536, 768);
  // 3) Fused attention
  attn_kernel<64, 256, 2, 4, 15, 3, 4, 4, 2><<<64 * 12, 256, 0, stream>>>(zx_q, zx_kv, zx_rpb, zx_o);
  attn_kernel<256, 64, 4, 2, 7, 4, 3, 8, 1><<<64 * 12, 256, 0, stream>>>(xz_q, xz_kv, xz_rpb, xz_o);
  // 4) Output projections + residual (f32 -> d_out)
  gemm_kernel<1,0><<<dim3(6, ZT/128), 256, 0, stream>>>(zx_o, wb_zx_pw, zx_pb, z, z2, ZT, 768, 768);
  gemm_kernel<1,0><<<dim3(6, XT/128), 256, 0, stream>>>(xz_o, wb_xz_pw, xz_pb, x, x2, XT, 768, 768);
  // 5) z MLP
  ln_kernel<<<(int)ZT, 256, 0, stream>>>(z2, z_ln2_g, z_ln2_b, z2n);
  gemm_kernel<2,1><<<dim3(24, ZT/128), 256, 0, stream>>>(z2n, wb_z_fc1, z_fc1_b, nullptr, zh, ZT, 3072, 768);
  gemm_kernel<1,0><<<dim3(6,  ZT/128), 256, 0, stream>>>(zh, wb_z_fc2, z_fc2_b, z2, z2, ZT, 768, 3072);
  // 6) x MLP
  ln_kernel<<<(int)XT, 256, 0, stream>>>(x2, x_ln2_g, x_ln2_b, x2n);
  gemm_kernel<2,1><<<dim3(24, XT/128), 256, 0, stream>>>(x2n, wb_x_fc1, x_fc1_b, nullptr, xh, XT, 3072, 768);
  gemm_kernel<1,0><<<dim3(6,  XT/128), 256, 0, stream>>>(xh, wb_x_fc2, x_fc2_b, x2, x2, XT, 768, 3072);
}
